// EIT_Localization_GNN_51788715655499
// MI455X (gfx1250) — matrix operand order repaired and checked by reading, not compiled, not run on hardware
//
#include <hip/hip_runtime.h>

#define NN 65536          // nodes
#define NE 524288         // edges
#define NG 256            // graphs
#define HD 128            // hidden dim
#define NPG 256           // nodes per graph (contiguous batch ids)
#define BNEPS 1e-5f

typedef __attribute__((ext_vector_type(16))) __bf16 v16bf;
typedef __attribute__((ext_vector_type(8)))  float  v8f;

union V16BF { v16bf v; unsigned int u[8]; };

__device__ __forceinline__ unsigned short f2bf(float f) {
  unsigned int u = __float_as_uint(f);
  u += 0x7FFFu + ((u >> 16) & 1u);          // round-to-nearest-even
  return (unsigned short)(u >> 16);
}
// order-preserving encode for float atomicMax on unsigned (0 == "empty")
__device__ __forceinline__ unsigned fenc(float f) {
  unsigned u = __float_as_uint(f);
  return (u & 0x80000000u) ? ~u : (u | 0x80000000u);
}
__device__ __forceinline__ float fdec(unsigned u) {
  return __uint_as_float((u & 0x80000000u) ? (u & 0x7FFFFFFFu) : ~u);
}

// ---------------- utility / conversion ----------------
__global__ void fill_u32_k(unsigned* p, unsigned v, int n) {
  int i = blockIdx.x * blockDim.x + threadIdx.x;
  if (i < n) p[i] = v;
}
__global__ void f32_to_bf16_k(const float* __restrict__ x, unsigned short* __restrict__ y, int n) {
  int i = blockIdx.x * blockDim.x + threadIdx.x;
  if (i < n) y[i] = f2bf(x[i]);
}
// Pack fp32 weight W[128,NC] into WMMA-B lane order so each lane reads its
// eight dwords for one k-step as ONE contiguous 32-byte chunk:
//   bp[(((ks*2 + half)*NC + n)*8 + j] = {bf16 W[ks*32+half*16+2j][n],
//                                        bf16 W[ks*32+half*16+2j+1][n]}
__global__ void pack_b_k(const float* __restrict__ W, unsigned* __restrict__ bp, int NC, int n) {
  int i = blockIdx.x * blockDim.x + threadIdx.x;
  if (i >= n) return;
  int j    = i & 7;
  int rest = i >> 3;
  int c    = rest % NC;
  int hh   = rest / NC;              // 0..7 == ks*2 + half
  int half = hh & 1, ks = hh >> 1;
  int k = ks * 32 + half * 16 + 2 * j;
  unsigned lo = f2bf(W[(size_t)k * NC + c]);
  unsigned hi = f2bf(W[(size_t)(k + 1) * NC + c]);
  bp[i] = lo | (hi << 16);
}

// ---------------- WMMA bf16 GEMM: C[M,NC] = A[M,128]@B (+ Cin) --------------
// One wave -> one 16x16 C tile, K=128 fully unrolled (4 x wmma 16x16x32).
// block = 256 threads = 8 waves covering 128 cols. grid = (M/16, NC/128).
// Straight-line body (template removes epilogue branches); EXEC all ones.
template <bool HASC>
__global__ __launch_bounds__(256)
void wmma_gemm_k(const unsigned short* __restrict__ A,   // bf16 row-major [M,128]
                 const unsigned int* __restrict__ Bp,    // packed weight (pack_b_k)
                 const float* __restrict__ Cin,
                 float* __restrict__ C,
                 int NC)
{
  const int lane = threadIdx.x & 31;
  const int wave = threadIdx.x >> 5;
  const int half = lane >> 4;          // 0: lanes 0-15, 1: lanes 16-31
  const int l16  = lane & 15;
  const int row0 = blockIdx.x * 16;
  const int col0 = blockIdx.y * 128 + wave * 16;
  const int col  = col0 + l16;

  const unsigned short* Arow = A + (size_t)(row0 + l16) * 128;
  v8f acc = {};
#pragma unroll
  for (int ks = 0; ks < 4; ++ks) {
    const int kb = ks * 32;
    V16BF a, b;
    // A (16-bit 16x32): lane holds row l16; VGPR j -> K=kb+half*16+2j
    // (contiguous, matches packed B). Two contiguous 16B loads.
    uint4 ra0 = *(const uint4*)(Arow + kb + half * 16);
    uint4 ra1 = *(const uint4*)(Arow + kb + half * 16 + 8);
    a.u[0]=ra0.x; a.u[1]=ra0.y; a.u[2]=ra0.z; a.u[3]=ra0.w;
    a.u[4]=ra1.x; a.u[5]=ra1.y; a.u[6]=ra1.z; a.u[7]=ra1.w;
    // B (16-bit 32x16): lane's 8 dwords pre-packed contiguously -> two 16B loads.
    const uint4* bp = (const uint4*)(Bp + ((size_t)(ks * 2 + half) * NC + col) * 8);
    uint4 rb0 = bp[0];
    uint4 rb1 = bp[1];
    b.u[0]=rb0.x; b.u[1]=rb0.y; b.u[2]=rb0.z; b.u[3]=rb0.w;
    b.u[4]=rb1.x; b.u[5]=rb1.y; b.u[6]=rb1.z; b.u[7]=rb1.w;
    acc = __builtin_amdgcn_wmma_f32_16x16x32_bf16(false, a.v, false, b.v,
                                                  (short)0, acc, false, false);
  }
  // C/D (32-bit 16x16): VGPR r -> row = row0 + half*8 + r, col = col0 + l16
  const size_t base = (size_t)(row0 + half * 8) * NC + col;
  float* __restrict__ cp = C + base;
  const float* __restrict__ ci = Cin + base;
#pragma unroll
  for (int r = 0; r < 8; ++r) {
    float v = acc[r];
    if (HASC) v += ci[(size_t)r * NC];
    cp[(size_t)r * NC] = v;
  }
}

// ---------------- small encoders ----------------
__global__ void node_enc_k(const float* __restrict__ x, const float* __restrict__ w,
                           const float* __restrict__ b, float* __restrict__ h) {
  int i = blockIdx.x * blockDim.x + threadIdx.x;      // NN*HD
  if (i >= NN * HD) return;
  int n = i >> 7, f = i & 127;
  float s = b[f];
  const float* xp = x + (size_t)n * 16;
#pragma unroll
  for (int k = 0; k < 16; ++k) s += xp[k] * w[k * HD + f];
  h[i] = fmaxf(s, 0.f);
}
__global__ void delta_enc_k(const float* __restrict__ dl, const float* __restrict__ w,
                            const float* __restrict__ b, float* __restrict__ out) {
  int g = blockIdx.x, f = threadIdx.x;                // NG blocks x 128
  float s = b[f];
  const float* dp = dl + (size_t)g * 208;
  for (int k = 0; k < 208; ++k) s += dp[k] * w[k * HD + f];
  out[g * HD + f] = fmaxf(s, 0.f);
}

// ---------------- graph conv scatter ----------------
__global__ void scatter_add_k(const int* __restrict__ src, const int* __restrict__ dst,
                              const float* __restrict__ h, float* __restrict__ agg) {
  int i = blockIdx.x * blockDim.x + threadIdx.x;      // NE*HD
  if (i >= NE * HD) return;
  int e = i >> 7, f = i & 127;
  atomicAdd(&agg[(size_t)dst[e] * HD + f], h[(size_t)src[e] * HD + f]);
}

// ---------------- batchnorm ----------------
__global__ void bn_stats_k(const float* __restrict__ X, float* mean, float* var) {
  int f = blockIdx.x, t = threadIdx.x;                // HD blocks x 256
  float s = 0.f, s2 = 0.f;
  for (int i = t; i < NN; i += 256) {
    float v = X[(size_t)i * HD + f];
    s += v; s2 += v * v;
  }
  __shared__ float sh[256], sh2[256];
  sh[t] = s; sh2[t] = s2; __syncthreads();
  for (int w = 128; w > 0; w >>= 1) {
    if (t < w) { sh[t] += sh[t + w]; sh2[t] += sh2[t + w]; }
    __syncthreads();
  }
  if (t == 0) {
    float m = sh[0] / (float)NN;
    mean[f] = m;
    var[f]  = sh2[0] / (float)NN - m * m;
  }
}
__global__ void bn_apply_relu_k(const float* __restrict__ X, const float* __restrict__ mean,
                                const float* __restrict__ var, const float* __restrict__ g,
                                const float* __restrict__ b, float* __restrict__ Y) {
  int i = blockIdx.x * blockDim.x + threadIdx.x;      // NN*HD
  if (i >= NN * HD) return;
  int f = i & 127;
  float v = (X[i] - mean[f]) * rsqrtf(var[f] + BNEPS) * g[f] + b[f];
  Y[i] = fmaxf(v, 0.f);
}

// ---------------- GAT ----------------
__global__ void attn_dots_k(const float* __restrict__ xh, const float* __restrict__ as,
                            const float* __restrict__ ad, float* __restrict__ aS,
                            float* __restrict__ aD) {
  int i = blockIdx.x * blockDim.x + threadIdx.x;      // NN*8
  if (i >= NN * 8) return;
  int n = i >> 3, h = (i >> 1) & 3, which = i & 1;
  const float* att = which ? ad : as;
  const float* xp = xh + (size_t)n * 512 + h * HD;
  float s = 0.f;
  for (int f = 0; f < HD; ++f) s += xp[f] * att[h * HD + f];
  (which ? aD : aS)[n * 4 + h] = s;
}
__global__ void gat_max_k(const int* __restrict__ src, const int* __restrict__ dst,
                          const float* __restrict__ aS, const float* __restrict__ aD,
                          unsigned* __restrict__ emax) {
  int i = blockIdx.x * blockDim.x + threadIdx.x;      // (NE+NN)*4
  if (i >= (NE + NN) * 4) return;
  int e = i >> 2, h = i & 3;
  int s = (e < NE) ? src[e] : (e - NE);
  int d = (e < NE) ? dst[e] : (e - NE);
  float v = aS[s * 4 + h] + aD[d * 4 + h];
  v = (v < 0.f) ? 0.2f * v : v;                        // leaky_relu(0.2)
  atomicMax(&emax[d * 4 + h], fenc(v));
}
__global__ void gat_exp_k(const int* __restrict__ src, const int* __restrict__ dst,
                          const float* __restrict__ aS, const float* __restrict__ aD,
                          const unsigned* __restrict__ emax, float* __restrict__ exb,
                          float* __restrict__ denom) {
  int i = blockIdx.x * blockDim.x + threadIdx.x;      // (NE+NN)*4
  if (i >= (NE + NN) * 4) return;
  int e = i >> 2, h = i & 3;
  int s = (e < NE) ? src[e] : (e - NE);
  int d = (e < NE) ? dst[e] : (e - NE);
  float v = aS[s * 4 + h] + aD[d * 4 + h];
  v = (v < 0.f) ? 0.2f * v : v;
  float ex = expf(v - fdec(emax[d * 4 + h]));
  exb[i] = ex;
  atomicAdd(&denom[d * 4 + h], ex);
}
__global__ void gat_msg_k(const int* __restrict__ src, const int* __restrict__ dst,
                          const float* __restrict__ xh, const float* __restrict__ exb,
                          const float* __restrict__ denom, float* __restrict__ out) {
  int i = blockIdx.x * blockDim.x + threadIdx.x;      // (NE+NN)*HD
  if (i >= (NE + NN) * HD) return;
  int e = i >> 7, f = i & 127;
  int s = (e < NE) ? src[e] : (e - NE);
  int d = (e < NE) ? dst[e] : (e - NE);
  float acc = 0.f;
#pragma unroll
  for (int h = 0; h < 4; ++h) {
    float alpha = exb[e * 4 + h] / (denom[d * 4 + h] + 1e-16f);
    acc += alpha * xh[(size_t)s * 512 + h * HD + f];
  }
  atomicAdd(&out[(size_t)d * HD + f], 0.25f * acc);    // 0.25 = mean over heads
}

// ---------------- pooling / layernorm / head ----------------
__global__ void pool_k(const float* __restrict__ h, float* __restrict__ rep) {
  int g = blockIdx.x, f = threadIdx.x;                // NG blocks x 128
  const float* hp = h + (size_t)g * NPG * HD + f;
  float s = 0.f, mx = -3.402823466e38f;
  for (int i = 0; i < NPG; ++i) {
    float v = hp[(size_t)i * HD];
    s += v; mx = fmaxf(mx, v);
  }
  rep[g * 2 * HD + f]      = s * (1.f / (float)NPG);
  rep[g * 2 * HD + HD + f] = mx;
}
__global__ void layernorm_k(const float* __restrict__ X, const float* __restrict__ g,
                            const float* __restrict__ b, float* __restrict__ Y, int D) {
  int row = blockIdx.x, t = threadIdx.x;              // block = D (<=256)
  __shared__ float sh[256];
  float v = X[(size_t)row * D + t];
  sh[t] = v; __syncthreads();
  for (int s = D >> 1; s > 0; s >>= 1) { if (t < s) sh[t] += sh[t + s]; __syncthreads(); }
  float mu = sh[0] / (float)D;
  __syncthreads();
  float dd = v - mu;
  sh[t] = dd * dd; __syncthreads();
  for (int s = D >> 1; s > 0; s >>= 1) { if (t < s) sh[t] += sh[t + s]; __syncthreads(); }
  float var = sh[0] / (float)D;
  Y[(size_t)row * D + t] = dd * rsqrtf(var + BNEPS) * g[t] + b[t];
}
__global__ void out_head_k(const float* __restrict__ grep, const float* __restrict__ drep,
                           const float* __restrict__ w, const float* __restrict__ b,
                           float* __restrict__ out) {
  int i = blockIdx.x * blockDim.x + threadIdx.x;      // NG*6
  if (i >= NG * 6) return;
  int g = i / 6, o = i - g * 6;
  float s = b[o];
  const float* gp = grep + (size_t)g * 2 * HD;
  const float* dp = drep + (size_t)g * HD;
  for (int k = 0; k < 2 * HD; ++k) s += gp[k] * w[k * 6 + o];
  for (int k = 0; k < HD; ++k)     s += dp[k] * w[(2 * HD + k) * 6 + o];
  out[i] = 1.f / (1.f + expf(-s));
}

static inline dim3 g1(long long n) { return dim3((unsigned)((n + 255) / 256)); }

extern "C" void kernel_launch(void* const* d_in, const int* in_sizes, int n_in,
                              void* d_out, int out_size, void* d_ws, size_t ws_size,
                              hipStream_t stream) {
  (void)in_sizes; (void)n_in; (void)out_size; (void)ws_size;
  const float* x       = (const float*)d_in[0];
  const float* delta   = (const float*)d_in[1];
  const int*   src     = (const int*)d_in[2];
  const int*   dst     = ((const int*)d_in[2]) + NE;
  /* d_in[3] = batch: graphs are contiguous NPG-node blocks, index unused */
  const float* w_node  = (const float*)d_in[4];
  const float* b_node  = (const float*)d_in[5];
  const float* w_delta = (const float*)d_in[6];
  const float* b_delta = (const float*)d_in[7];
  const float* w1_rel  = (const float*)d_in[8];
  /* b1_rel (d_in[9]) cancels exactly in the following BatchNorm */
  const float* w1_root = (const float*)d_in[10];
  const float* w_gat   = (const float*)d_in[11];
  const float* att_src = (const float*)d_in[12];
  const float* att_dst = (const float*)d_in[13];
  /* b_gat (d_in[14]) cancels exactly in BatchNorm */
  const float* w3_rel  = (const float*)d_in[15];
  /* b3_rel (d_in[16]) cancels exactly in BatchNorm */
  const float* w3_root = (const float*)d_in[17];
  const float* g_bn1 = (const float*)d_in[18]; const float* b_bn1 = (const float*)d_in[19];
  const float* g_bn2 = (const float*)d_in[20]; const float* b_bn2 = (const float*)d_in[21];
  const float* g_bn3 = (const float*)d_in[22]; const float* b_bn3 = (const float*)d_in[23];
  const float* g_gn  = (const float*)d_in[24]; const float* b_gn  = (const float*)d_in[25];
  const float* g_dn  = (const float*)d_in[26]; const float* b_dn  = (const float*)d_in[27];
  const float* w_out = (const float*)d_in[28]; const float* b_out = (const float*)d_in[29];
  float* out = (float*)d_out;

  // ---- workspace carve ----
  char* p = (char*)d_ws;
  auto alloc = [&](size_t bytes) { void* r = (void*)p; p += (bytes + 255) & ~(size_t)255; return r; };
  float*          hA    = (float*)alloc(sizeof(float) * NN * HD);       // activations
  float*          hB    = (float*)alloc(sizeof(float) * NN * HD);       // pre-BN tmp
  float*          agg   = (float*)alloc(sizeof(float) * NN * HD);       // scatter sum
  float*          xh    = (float*)alloc(sizeof(float) * NN * 4 * HD);   // GAT transformed
  float*          aS    = (float*)alloc(sizeof(float) * NN * 4);
  float*          aD    = (float*)alloc(sizeof(float) * NN * 4);
  float*          denom = (float*)alloc(sizeof(float) * NN * 4);
  unsigned*       emax  = (unsigned*)alloc(sizeof(unsigned) * NN * 4);
  float*          exb   = (float*)alloc(sizeof(float) * (NE + NN) * 4);
  unsigned short* hbf   = (unsigned short*)alloc(sizeof(unsigned short) * NN * HD);
  unsigned*       bpack = (unsigned*)alloc(sizeof(unsigned) * (HD / 2) * 4 * HD);
  float*          meanv = (float*)alloc(sizeof(float) * HD);
  float*          varv  = (float*)alloc(sizeof(float) * HD);
  float*          grep  = (float*)alloc(sizeof(float) * NG * 2 * HD);
  float*          grepL = (float*)alloc(sizeof(float) * NG * 2 * HD);
  float*          drep  = (float*)alloc(sizeof(float) * NG * HD);
  float*          drepL = (float*)alloc(sizeof(float) * NG * HD);

  const int NH  = NN * HD;
  const int EW  = NE * HD;
  const int EN4 = (NE + NN) * 4;
  const int ENH = (NE + NN) * HD;

  // 1) node encoder
  node_enc_k<<<g1(NH), 256, 0, stream>>>(x, w_node, b_node, hA);

  // GraphConv + BN + relu (hA -> hA)
  auto graph_conv = [&](const float* wrel, const float* wroot,
                        const float* gbn, const float* bbn) {
    fill_u32_k<<<g1(NH), 256, 0, stream>>>((unsigned*)agg, 0u, NH);
    scatter_add_k<<<g1(EW), 256, 0, stream>>>(src, dst, hA, agg);
    f32_to_bf16_k<<<g1(NH), 256, 0, stream>>>(agg, hbf, NH);
    pack_b_k<<<g1((HD / 2) * HD), 256, 0, stream>>>(wrel, bpack, HD, (HD / 2) * HD);
    wmma_gemm_k<false><<<dim3(NN / 16, 1), 256, 0, stream>>>(hbf, bpack, hB, hB, HD);
    f32_to_bf16_k<<<g1(NH), 256, 0, stream>>>(hA, hbf, NH);
    pack_b_k<<<g1((HD / 2) * HD), 256, 0, stream>>>(wroot, bpack, HD, (HD / 2) * HD);
    wmma_gemm_k<true><<<dim3(NN / 16, 1), 256, 0, stream>>>(hbf, bpack, hB, hB, HD);
    bn_stats_k<<<HD, 256, 0, stream>>>(hB, meanv, varv);
    bn_apply_relu_k<<<g1(NH), 256, 0, stream>>>(hB, meanv, varv, gbn, bbn, hA);
  };

  // 2) GraphConv 1
  graph_conv(w1_rel, w1_root, g_bn1, b_bn1);

  // 3) GATConv + BN + relu
  f32_to_bf16_k<<<g1(NH), 256, 0, stream>>>(hA, hbf, NH);
  pack_b_k<<<g1((HD / 2) * 4 * HD), 256, 0, stream>>>(w_gat, bpack, 4 * HD, (HD / 2) * 4 * HD);
  wmma_gemm_k<false><<<dim3(NN / 16, 4), 256, 0, stream>>>(hbf, bpack, xh, xh, 4 * HD);
  attn_dots_k<<<g1(NN * 8), 256, 0, stream>>>(xh, att_src, att_dst, aS, aD);
  fill_u32_k<<<g1(NN * 4), 256, 0, stream>>>(emax, 0u, NN * 4);
  fill_u32_k<<<g1(NN * 4), 256, 0, stream>>>((unsigned*)denom, 0u, NN * 4);
  gat_max_k<<<g1(EN4), 256, 0, stream>>>(src, dst, aS, aD, emax);
  gat_exp_k<<<g1(EN4), 256, 0, stream>>>(src, dst, aS, aD, emax, exb, denom);
  fill_u32_k<<<g1(NH), 256, 0, stream>>>((unsigned*)hB, 0u, NH);
  gat_msg_k<<<g1(ENH), 256, 0, stream>>>(src, dst, xh, exb, denom, hB);
  bn_stats_k<<<HD, 256, 0, stream>>>(hB, meanv, varv);
  bn_apply_relu_k<<<g1(NH), 256, 0, stream>>>(hB, meanv, varv, g_bn2, b_bn2, hA);

  // 4) GraphConv 3
  graph_conv(w3_rel, w3_root, g_bn3, b_bn3);

  // 5) pooling + layernorm
  pool_k<<<NG, HD, 0, stream>>>(hA, grep);
  layernorm_k<<<NG, 2 * HD, 0, stream>>>(grep, g_gn, b_gn, grepL, 2 * HD);

  // 6) delta encoder + layernorm
  delta_enc_k<<<NG, HD, 0, stream>>>(delta, w_delta, b_delta, drep);
  layernorm_k<<<NG, HD, 0, stream>>>(drep, g_dn, b_dn, drepL, HD);

  // 7) output head
  out_head_k<<<g1(NG * 6), 256, 0, stream>>>(grepL, drepL, w_out, b_out, out);
}
